// Encoder_7_3710851744012
// MI455X (gfx1250) — matrix operand order repaired and checked by reading, not compiled, not run on hardware
//
#include <hip/hip_runtime.h>

typedef unsigned short u16;
typedef __attribute__((ext_vector_type(16))) __bf16 v16bf;
typedef __attribute__((ext_vector_type(8)))  float  v8f;

union ABfrag { uint4 q[2]; v16bf v; };

__device__ __forceinline__ u16 f2bf(float f) {
    union { float f; unsigned u; } v; v.f = f;
    unsigned u = v.u;
    unsigned r = (u + 0x7FFFu + ((u >> 16) & 1u)) >> 16;
    return (u16)r;
}
__device__ __forceinline__ float sigf(float x) { return 1.f / (1.f + __expf(-x)); }

// ---------------------------------------------------------------------------
// bf16 WMMA GEMM, 2x2 register-blocked: each wave computes a 32(co) x 32(t)
// region (4 accumulators). Per K-chunk of 32: 2 A-frags + 2 B-frags (8 b128
// loads) feed 4 v_wmma_f32_16x16x32_bf16 -> 2 loads/WMMA, A/B reused 2x.
// out[b][t][co] = bias[co] + sum_k sum_ci W[co][k][ci] * act[b][t+rowOff+k][ci]
// act rows: [b][tp][ci], tp in [0,196), row stride rstride, kLen % 32 == 0.
// ---------------------------------------------------------------------------
__global__ __launch_bounds__(256) void wmma_gemm_kernel(
    const u16* __restrict__ act, const u16* __restrict__ wts,
    const float* __restrict__ bias, float* __restrict__ out,
    int Mtiles, int kLen, int rstride, int taps, int rowOff, int Cout)
{
    const int NG = 6;                       // 192 / 32 N-groups
    int wave = blockIdx.x * 8 + (threadIdx.x >> 5);
    int lane = threadIdx.x & 31;
    int Mg   = Mtiles >> 1;
    int perB = Mg * NG;
    int b    = wave / perB;
    if (b >= 256) return;                   // wave-uniform, EXEC stays full
    int r  = wave % perB;
    int mg = r / NG;
    int ng = r % NG;
    int l15 = lane & 15;
    int hi  = lane >> 4;
    int t0  = ng * 32 + l15;

    // A fragments: rows mg*32 + l15 (+16); per-lane contiguous K run of 16 at (hi?16:0)
    const u16* w0 = wts + (size_t)(mg * 32 + l15) * (size_t)(taps * kLen) + hi * 16;
    const u16* w1 = w0 + (size_t)16 * (size_t)(taps * kLen);
    // B fragments: cols t0 (+16); per-lane contiguous K run of 16 at (hi?16:0)
    const u16* b0 = act + ((size_t)b * 196 + rowOff + t0) * (size_t)rstride + hi * 16;
    const u16* b1 = b0 + (size_t)16 * rstride;

    v8f c00, c01, c10, c11;
#pragma unroll
    for (int i = 0; i < 8; i++) { c00[i] = 0.f; c01[i] = 0.f; c10[i] = 0.f; c11[i] = 0.f; }

    for (int k = 0; k < taps; k++) {
        const u16* wp0 = w0 + (size_t)k * kLen;
        const u16* wp1 = w1 + (size_t)k * kLen;
        const u16* ap0 = b0 + (size_t)k * rstride;
        const u16* ap1 = b1 + (size_t)k * rstride;
        if (k + 1 < taps) {                 // prefetch next tap's activation rows
            __builtin_prefetch(ap0 + rstride, 0, 1);
            __builtin_prefetch(ap1 + rstride, 0, 1);
        }
        for (int c0 = 0; c0 < kLen; c0 += 32) {
            ABfrag A0, A1, B0, B1;
            A0.q[0] = *(const uint4*)(wp0 + c0);
            A0.q[1] = *(const uint4*)(wp0 + c0 + 8);
            A1.q[0] = *(const uint4*)(wp1 + c0);
            A1.q[1] = *(const uint4*)(wp1 + c0 + 8);
            B0.q[0] = *(const uint4*)(ap0 + c0);
            B0.q[1] = *(const uint4*)(ap0 + c0 + 8);
            B1.q[0] = *(const uint4*)(ap1 + c0);
            B1.q[1] = *(const uint4*)(ap1 + c0 + 8);
            c00 = __builtin_amdgcn_wmma_f32_16x16x32_bf16(false, A0.v, false, B0.v, (short)0, c00, false, false);
            c01 = __builtin_amdgcn_wmma_f32_16x16x32_bf16(false, A0.v, false, B1.v, (short)0, c01, false, false);
            c10 = __builtin_amdgcn_wmma_f32_16x16x32_bf16(false, A1.v, false, B0.v, (short)0, c10, false, false);
            c11 = __builtin_amdgcn_wmma_f32_16x16x32_bf16(false, A1.v, false, B1.v, (short)0, c11, false, false);
        }
    }

    // D layout per tile: lane col N=t ; VGPR r -> row M = hi*8 + r
    int coA = mg * 32 + hi * 8;             // M-tile 0
    int coB = coA + 16;                     // M-tile 1
    float4 bA0 = *(const float4*)(bias + coA);
    float4 bA1 = *(const float4*)(bias + coA + 4);
    float4 bB0 = *(const float4*)(bias + coB);
    float4 bB1 = *(const float4*)(bias + coB + 4);

    float* r00 = out + ((size_t)b * 192 + t0)      * (size_t)Cout + coA;  // (m0,n0)
    float* r01 = out + ((size_t)b * 192 + t0 + 16) * (size_t)Cout + coA;  // (m0,n1)
    float* r10 = out + ((size_t)b * 192 + t0)      * (size_t)Cout + coB;  // (m1,n0)
    float* r11 = out + ((size_t)b * 192 + t0 + 16) * (size_t)Cout + coB;  // (m1,n1)

    float4 o;
    o.x=c00[0]+bA0.x; o.y=c00[1]+bA0.y; o.z=c00[2]+bA0.z; o.w=c00[3]+bA0.w; *(float4*)(r00)     = o;
    o.x=c00[4]+bA1.x; o.y=c00[5]+bA1.y; o.z=c00[6]+bA1.z; o.w=c00[7]+bA1.w; *(float4*)(r00 + 4) = o;
    o.x=c01[0]+bA0.x; o.y=c01[1]+bA0.y; o.z=c01[2]+bA0.z; o.w=c01[3]+bA0.w; *(float4*)(r01)     = o;
    o.x=c01[4]+bA1.x; o.y=c01[5]+bA1.y; o.z=c01[6]+bA1.z; o.w=c01[7]+bA1.w; *(float4*)(r01 + 4) = o;
    o.x=c10[0]+bB0.x; o.y=c10[1]+bB0.y; o.z=c10[2]+bB0.z; o.w=c10[3]+bB0.w; *(float4*)(r10)     = o;
    o.x=c10[4]+bB1.x; o.y=c10[5]+bB1.y; o.z=c10[6]+bB1.z; o.w=c10[7]+bB1.w; *(float4*)(r10 + 4) = o;
    o.x=c11[0]+bB0.x; o.y=c11[1]+bB0.y; o.z=c11[2]+bB0.z; o.w=c11[3]+bB0.w; *(float4*)(r11)     = o;
    o.x=c11[4]+bB1.x; o.y=c11[5]+bB1.y; o.z=c11[6]+bB1.z; o.w=c11[7]+bB1.w; *(float4*)(r11 + 4) = o;
}

// ---------------------------------------------------------------------------
// Weight packing: W (Cout, Cin, taps) f32 -> bf16 [co][k][ci_pad] (zeros pad)
// ---------------------------------------------------------------------------
__global__ void pack_w_kernel(const float* __restrict__ w, u16* __restrict__ dst,
                              int Cout, int Cin, int taps, int kpad)
{
    int n = blockIdx.x * 256 + threadIdx.x;
    int tot = Cout * taps * kpad;
    if (n >= tot) return;
    int co = n / (taps * kpad);
    int r  = n % (taps * kpad);
    int k  = r / kpad;
    int ci = r % kpad;
    float v = (ci < Cin) ? w[((size_t)co * Cin + ci) * taps + k] : 0.f;
    dst[(size_t)co * taps * kpad + (size_t)k * kpad + ci] = f2bf(v);
}

// Layer-1 input: x_f0 (B,337,192) f32 channel-major -> time-major bf16 buffers
__global__ void input_transpose(const float* __restrict__ xf0,
                                u16* __restrict__ xb, u16* __restrict__ f0b)
{
    int n = blockIdx.x * 256 + threadIdx.x;
    if (n >= 256 * 337 * 192) return;
    int t = n % 192;
    int rest = n / 192;
    int ci = rest % 337;
    int b  = rest / 337;
    u16 v = f2bf(xf0[n]);
    if (ci < 80) xb [((size_t)b * 196 + 2 + t) * 512 + ci]        = v;
    else         f0b[((size_t)b * 196 + 2 + t) * 288 + (ci - 80)] = v;
}

// ---------------------------------------------------------------------------
// GroupNorm: stats (mean, inv_std) per (b, group of 16 ch) over 16ch x 192t
// ---------------------------------------------------------------------------
__global__ __launch_bounds__(256) void gn_stats(const float* __restrict__ y,
                                                float* __restrict__ stats, int C, int G)
{
    __shared__ float s1[256], s2[256];
    int g = blockIdx.x % G;
    int b = blockIdx.x / G;
    int tid = threadIdx.x;
    float a = 0.f, q = 0.f;
    for (int i = tid; i < 192 * 16; i += 256) {
        int t = i >> 4, c = g * 16 + (i & 15);
        float v = y[((size_t)b * 192 + t) * C + c];
        a += v; q += v * v;
    }
    s1[tid] = a; s2[tid] = q;
    __syncthreads();
    for (int st = 128; st > 0; st >>= 1) {
        if (tid < st) { s1[tid] += s1[tid + st]; s2[tid] += s2[tid + st]; }
        __syncthreads();
    }
    if (tid == 0) {
        float m   = s1[0] / 3072.f;
        float var = s2[0] / 3072.f - m * m;
        stats[(size_t)blockIdx.x * 2 + 0] = m;
        stats[(size_t)blockIdx.x * 2 + 1] = rsqrtf(var + 1e-5f);
    }
}

__global__ void gn_apply(float* __restrict__ y, const float* __restrict__ stats,
                         const float* __restrict__ gamma, const float* __restrict__ beta,
                         int C, int G)
{
    size_t n = (size_t)blockIdx.x * 256 + threadIdx.x;
    int c = (int)(n % C);
    size_t bt = n / C;
    int b = (int)(bt / 192);
    int g = c >> 4;
    const float* st = stats + ((size_t)b * G + g) * 2;
    float v = (y[n] - st[0]) * st[1] * gamma[c] + beta[c];
    y[n] = v > 0.f ? v : 0.f;
}

// ---------------------------------------------------------------------------
// interp_lnr stage 1: per-batch serial scan -> compacted (srcIdx, lam, count)
// ---------------------------------------------------------------------------
__global__ void interp_prep(const float* __restrict__ scales, const int* __restrict__ lens,
                            int layer, int* __restrict__ sidx, float* __restrict__ slam,
                            int* __restrict__ cnt)
{
    int b = threadIdx.x;                     // 256 threads, 1 block
    int base = layer * 1792 + b * 7;         // (3, B*K), K=7
    float offs = 0.f;
    int p = 0;
    for (int j = 0; j < 7; j++) {
        float sc = scales[base + j];
        int len  = lens[base + j];
        float lenm1 = (float)len - 1.f;
        for (int jj = 0; jj < 64; jj++) {
            float is  = (float)jj / sc;
            float fl  = floorf(is);
            float lam = is - fl;
            float io  = fl + offs;
            if (fl < lenm1 && io < 191.f) {
                if (p < 192) {
                    int ib = (int)io;
                    if (ib < 0) ib = 0;
                    if (ib > 190) ib = 190;
                    sidx[b * 192 + p] = ib;
                    slam[b * 192 + p] = lam;
                }
                p++;
            }
        }
        offs += (float)len;
    }
    cnt[b] = p < 192 ? p : 192;
}

// interp_lnr stage 2: gather + lerp full channel rows, emit bf16 into padded conv input
__global__ void interp_write(const float* __restrict__ y, const int* __restrict__ sidx,
                             const float* __restrict__ slam, const int* __restrict__ cnt,
                             u16* __restrict__ dst, int C, int Cpad, int dstStride)
{
    int b = blockIdx.x / 192;
    int p = blockIdx.x % 192;
    int cn = cnt[b];
    bool valid = p < cn;
    int idx = 0; float lam = 0.f;
    if (valid) { idx = sidx[b * 192 + p]; lam = slam[b * 192 + p]; }
    u16* drow = dst + ((size_t)b * 196 + 2 + p) * dstStride;
    const float* r0 = y + ((size_t)b * 192 + idx) * C;
    for (int c = threadIdx.x; c < Cpad; c += 256) {
        float v = 0.f;
        if (valid && c < C) v = (1.f - lam) * r0[c] + lam * r0[C + c];
        drow[c] = f2bf(v);
    }
}

// ---------------------------------------------------------------------------
// LSTM recurrence: one wave per batch; lane l < H owns hidden index l
// (gate rows l, l+H, l+2H, l+3H). h shared via per-wave LDS.
// ---------------------------------------------------------------------------
__global__ __launch_bounds__(256) void lstm_recur(const float* __restrict__ pre,
                                                  const float* __restrict__ whh,
                                                  float* __restrict__ hcat,
                                                  int H, int reverse, int outOff)
{
    __shared__ float hbuf[8][32];
    int w = threadIdx.x >> 5, lane = threadIdx.x & 31;
    int b = blockIdx.x * 8 + w;
    float cc = 0.f;
    hbuf[w][lane] = 0.f;
    __syncthreads();
    for (int s = 0; s < 192; s++) {
        int t = reverse ? (191 - s) : s;
        float hh = 0.f;
        if (lane < H) {
            const float* pr = pre + ((size_t)b * 192 + t) * 4 * H;
            float zi = pr[lane], zf = pr[lane + H], zg = pr[lane + 2 * H], zo = pr[lane + 3 * H];
            for (int j = 0; j < H; j++) {
                float hj = hbuf[w][j];
                zi += whh[(size_t)(lane        ) * H + j] * hj;
                zf += whh[(size_t)(lane +     H) * H + j] * hj;
                zg += whh[(size_t)(lane + 2 * H) * H + j] * hj;
                zo += whh[(size_t)(lane + 3 * H) * H + j] * hj;
            }
            cc = sigf(zf) * cc + sigf(zi) * tanhf(zg);
            hh = sigf(zo) * tanhf(cc);
            hcat[((size_t)b * 192 + t) * (2 * H) + outOff + lane] = hh;
        }
        __syncthreads();
        if (lane < H) hbuf[w][lane] = hh;
        __syncthreads();
    }
}

// Small gate pre-projection for LSTM1 layer 2 (K = 16, both directions)
__global__ void lstm_pre_small(const float* __restrict__ xin,
                               const float* __restrict__ wf, const float* __restrict__ bf,
                               const float* __restrict__ wb, const float* __restrict__ bb,
                               float* __restrict__ preF, float* __restrict__ preB)
{
    int n = blockIdx.x * 256 + threadIdx.x;    // B*192*2dirs*32gates
    int g = n & 31;
    int d = (n >> 5) & 1;
    int bt = n >> 6;
    if (bt >= 256 * 192) return;
    const float* W  = d ? wb : wf;
    const float* bi = d ? bb : bf;
    float s = bi[g];
    const float* xr = xin + (size_t)bt * 16;
    for (int j = 0; j < 16; j++) s += xr[j] * W[g * 16 + j];
    (d ? preB : preF)[(size_t)bt * 32 + g] = s;
}

// Final strided slice/concat -> d_out
__global__ void codes_kernel(const float* __restrict__ h2, const float* __restrict__ hf,
                             float* __restrict__ out)
{
    int n = blockIdx.x * 256 + threadIdx.x;
    if (n < 98304) {                                   // codes_x (B,24,16)
        int c = n & 15; int r = n >> 4; int t = r % 24; int b = r / 24;
        int tt = (c < 8) ? (8 * t + 7) : (8 * t);
        out[n] = h2[((size_t)b * 192 + tt) * 16 + c];
    } else if (n < 491520) {                           // codes_f0 (B,24,64)
        int m = n - 98304;
        int c = m & 63; int r = m >> 6; int t = r % 24; int b = r / 24;
        int tt = (c < 32) ? (8 * t + 7) : (8 * t);
        out[n] = hf[((size_t)b * 192 + tt) * 64 + c];
    }
}

// ---------------------------------------------------------------------------
// Host side
// ---------------------------------------------------------------------------
static inline int cdiv(long long a, long long b) { return (int)((a + b - 1) / b); }

extern "C" void kernel_launch(void* const* d_in, const int* in_sizes, int n_in,
                              void* d_out, int out_size, void* d_ws, size_t ws_size,
                              hipStream_t stream)
{
    (void)in_sizes; (void)n_in; (void)out_size; (void)ws_size;
    char* ws = (char*)d_ws;

    // Workspace layout (all offsets 256B aligned)
    const size_t SZ_XB  = 256ull * 196 * 512 * 2;   // bf16 x activations (t-major, padded)
    const size_t SZ_F0B = 256ull * 196 * 288 * 2;   // bf16 f0 activations
    const size_t SZ_YX  = 256ull * 192 * 512 * 4;   // f32 conv-x output [b][t][co]
    const size_t SZ_YF0 = 256ull * 192 * 256 * 4;   // f32 conv-f0 output
    const size_t SZ_WP  = 3989504ull * 2;           // packed bf16 weights

    const size_t OFF_XB   = 0;
    const size_t OFF_F0B  = OFF_XB  + SZ_XB;
    const size_t OFF_YX   = OFF_F0B + SZ_F0B;
    const size_t OFF_YF0  = OFF_YX  + SZ_YX;
    const size_t OFF_WP   = OFF_YF0 + SZ_YF0;
    const size_t OFF_STX  = OFF_WP  + SZ_WP;
    const size_t OFF_STF  = OFF_STX + 256ull * 32 * 2 * 4;
    const size_t OFF_SIDX = OFF_STF + 256ull * 16 * 2 * 4;
    const size_t OFF_SLAM = OFF_SIDX + 256ull * 192 * 4;
    const size_t OFF_CNT  = OFF_SLAM + 256ull * 192 * 4;

    u16*   xb     = (u16*)(ws + OFF_XB);
    u16*   f0b    = (u16*)(ws + OFF_F0B);
    float* yx     = (float*)(ws + OFF_YX);
    float* yf0    = (float*)(ws + OFF_YF0);
    u16*   wp     = (u16*)(ws + OFF_WP);
    float* statsX = (float*)(ws + OFF_STX);
    float* statsF = (float*)(ws + OFF_STF);
    int*   sidx   = (int*)(ws + OFF_SIDX);
    float* slam   = (float*)(ws + OFF_SLAM);
    int*   cnt    = (int*)(ws + OFF_CNT);

    // LSTM buffers alias conv-output regions (dead after layer-3 interp)
    float* preF   = (float*)(ws + OFF_YX);
    float* preB   = (float*)(ws + OFF_YX + 25165824ull);
    float* hcat1  = (float*)(ws + OFF_YF0);
    float* hcat2  = (float*)(ws + OFF_YF0 + 3145728ull);
    float* hcatF  = (float*)(ws + OFF_YF0 + 6291456ull);

    const float* scales = (const float*)d_in[43];
    const int*   lenseg = (const int*)d_in[44];

    // Zero activation buffers (covers time-pad rows + channel-pad columns)
    hipMemsetAsync(xb,  0, SZ_XB,  stream);
    hipMemsetAsync(f0b, 0, SZ_F0B, stream);

    // Pack all matmul weights to bf16
    const int wo_cx[3] = { 0, 245760, 1556480 };
    const int wo_cf[3] = { 2867200, 3235840, 3563520 };
    const int WO_L1F = 3891200, WO_L1B = 3907584, WO_L2F = 3923968, WO_L2B = 3956736;
    const int kx[3]   = { 96, 512, 512 };
    const int kf[3]   = { 288, 256, 256 };
    const int cinx[3] = { 80, 512, 512 };
    const int cinf[3] = { 257, 256, 256 };
    for (int i = 0; i < 3; i++) {
        pack_w_kernel<<<cdiv(512LL * 5 * kx[i], 256), 256, 0, stream>>>(
            (const float*)d_in[1 + 4 * i], wp + wo_cx[i], 512, cinx[i], 5, kx[i]);
        pack_w_kernel<<<cdiv(256LL * 5 * kf[i], 256), 256, 0, stream>>>(
            (const float*)d_in[13 + 4 * i], wp + wo_cf[i], 256, cinf[i], 5, kf[i]);
    }
    pack_w_kernel<<<cdiv(32LL * 512, 256), 256, 0, stream>>>((const float*)d_in[25], wp + WO_L1F, 32, 512, 1, 512);
    pack_w_kernel<<<cdiv(32LL * 512, 256), 256, 0, stream>>>((const float*)d_in[28], wp + WO_L1B, 32, 512, 1, 512);
    pack_w_kernel<<<cdiv(128LL * 256, 256), 256, 0, stream>>>((const float*)d_in[37], wp + WO_L2F, 128, 256, 1, 256);
    pack_w_kernel<<<cdiv(128LL * 256, 256), 256, 0, stream>>>((const float*)d_in[40], wp + WO_L2B, 128, 256, 1, 256);

    // Layer-1 input transpose to time-major bf16
    input_transpose<<<cdiv(256LL * 337 * 192, 256), 256, 0, stream>>>(
        (const float*)d_in[0], xb, f0b);

    // Three conv -> GN -> ReLU -> interp rounds
    for (int L = 0; L < 3; L++) {
        // conv x: 256 batches * (32/2) M-groups * 6 N-groups waves / 8 per block
        wmma_gemm_kernel<<<256 * 16 * 6 / 8, 256, 0, stream>>>(
            xb, wp + wo_cx[L], (const float*)d_in[2 + 4 * L], yx, 32, kx[L], 512, 5, 0, 512);
        // conv f0
        wmma_gemm_kernel<<<256 * 8 * 6 / 8, 256, 0, stream>>>(
            f0b, wp + wo_cf[L], (const float*)d_in[14 + 4 * L], yf0, 16, kf[L], 288, 5, 0, 256);

        gn_stats<<<256 * 32, 256, 0, stream>>>(yx, statsX, 512, 32);
        gn_apply<<<cdiv(256LL * 192 * 512, 256), 256, 0, stream>>>(
            yx, statsX, (const float*)d_in[3 + 4 * L], (const float*)d_in[4 + 4 * L], 512, 32);
        gn_stats<<<256 * 16, 256, 0, stream>>>(yf0, statsF, 256, 16);
        gn_apply<<<cdiv(256LL * 192 * 256, 256), 256, 0, stream>>>(
            yf0, statsF, (const float*)d_in[15 + 4 * L], (const float*)d_in[16 + 4 * L], 256, 16);

        interp_prep<<<1, 256, 0, stream>>>(scales, lenseg, L, sidx, slam, cnt);
        interp_write<<<256 * 192, 256, 0, stream>>>(yx,  sidx, slam, cnt, xb,  512, 512, 512);
        interp_write<<<256 * 192, 256, 0, stream>>>(yf0, sidx, slam, cnt, f0b, 256, 288, 288);
    }

    // ---- BiLSTM 1, layer 1 (in=512, H=8): WMMA gate projection from xb ----
    wmma_gemm_kernel<<<256 * 1 * 6 / 8, 256, 0, stream>>>(
        xb, wp + WO_L1F, (const float*)d_in[27], preF, 2, 512, 512, 1, 2, 32);
    wmma_gemm_kernel<<<256 * 1 * 6 / 8, 256, 0, stream>>>(
        xb, wp + WO_L1B, (const float*)d_in[30], preB, 2, 512, 512, 1, 2, 32);
    lstm_recur<<<32, 256, 0, stream>>>(preF, (const float*)d_in[26], hcat1, 8, 0, 0);
    lstm_recur<<<32, 256, 0, stream>>>(preB, (const float*)d_in[29], hcat1, 8, 1, 8);

    // ---- BiLSTM 1, layer 2 (in=16, H=8): tiny VALU projection ----
    lstm_pre_small<<<cdiv(256LL * 192 * 64, 256), 256, 0, stream>>>(
        hcat1, (const float*)d_in[31], (const float*)d_in[33],
        (const float*)d_in[34], (const float*)d_in[36], preF, preB);
    lstm_recur<<<32, 256, 0, stream>>>(preF, (const float*)d_in[32], hcat2, 8, 0, 0);
    lstm_recur<<<32, 256, 0, stream>>>(preB, (const float*)d_in[35], hcat2, 8, 1, 8);

    // ---- BiLSTM 2 (in=256, H=32): WMMA gate projection from f0b ----
    wmma_gemm_kernel<<<256 * 4 * 6 / 8, 256, 0, stream>>>(
        f0b, wp + WO_L2F, (const float*)d_in[39], preF, 8, 256, 288, 1, 2, 128);
    wmma_gemm_kernel<<<256 * 4 * 6 / 8, 256, 0, stream>>>(
        f0b, wp + WO_L2B, (const float*)d_in[42], preB, 8, 256, 288, 1, 2, 128);
    lstm_recur<<<32, 256, 0, stream>>>(preF, (const float*)d_in[38], hcatF, 32, 0, 0);
    lstm_recur<<<32, 256, 0, stream>>>(preB, (const float*)d_in[41], hcatF, 32, 1, 32);

    // ---- codes_x / codes_f0 -> d_out ----
    codes_kernel<<<cdiv(491520LL, 256), 256, 0, stream>>>(hcat2, hcatF, (float*)d_out);
}
